// CopyDecoder_40518721470647
// MI455X (gfx1250) — compile-verified, matrix operand order checked by reading, not run on hardware
//
#include <hip/hip_runtime.h>
#include <math.h>

#define BATCH   32
#define SRCLEN  400
#define TRGLEN  33
#define TSTEPS  32
#define HID     512
#define EMBD    512
#define NVOCAB  32000
#define NVEXT   32050
#define KFEAT   1536   // 3*HID
#define G4      2048   // 4*HID
#define UNKTOK  1

typedef __attribute__((ext_vector_type(16))) __bf16 v16bf;
typedef __attribute__((ext_vector_type(8)))  float  v8f;

static __device__ __forceinline__ unsigned short f2bf(float f) {
  union { float f; unsigned int u; } v; v.f = f;
  unsigned int u = v.u;
  unsigned int r = u + 0x7FFFu + ((u >> 16) & 1u);   // round-to-nearest-even
  return (unsigned short)(r >> 16);
}
static __device__ __forceinline__ __bf16 bfbits(unsigned short u) {
  __bf16 b; __builtin_memcpy(&b, &u, 2); return b;
}
static __device__ __forceinline__ float sigm(float x) { return 1.0f / (1.0f + __expf(-x)); }

static __device__ __forceinline__ v8f wmma_bf16(v16bf a, v16bf b, v8f c) {
  return __builtin_amdgcn_wmma_f32_16x16x32_bf16(false, a, false, b, (short)0, c, false, false);
}

// ---------------------------------------------------------------- converts
__global__ void k_f32_to_bf16(const float* __restrict__ src,
                              unsigned short* __restrict__ dst, int n) {
  int i = blockIdx.x * blockDim.x + threadIdx.x;
  if (i < n) dst[i] = f2bf(src[i]);
}

// ------------------------------------------------- enc_proj = enc @ W_attn^T
// A: enc_bf16 [12800 x 512] row-major (K contiguous)
// B: W_attn bf16 [512(h) x 512(e)] row-major; each row is one B column
// D: enc_proj fp32 [12800 x 512]
__global__ void k_encproj_gemm(const unsigned short* __restrict__ A,
                               const unsigned short* __restrict__ Wb,
                               float* __restrict__ D) {
  int wave = threadIdx.x >> 5, lane = threadIdx.x & 31;
  int w  = blockIdx.x * 8 + wave;        // 0..12799 wave-tiles
  int mt = w >> 5;                       // 400 tiles of M=32
  int nt = w & 31;                       // 32 tiles of N=16
  int m0 = mt * 32, n0 = nt * 16;
  int row = lane & 15;
  int kb  = (lane >> 4) << 3;
  const unsigned short* a0p = A  + (size_t)(m0 + row)      * 512;
  const unsigned short* a1p = A  + (size_t)(m0 + 16 + row) * 512;
  const unsigned short* bp  = Wb + (size_t)(n0 + row)      * 512;
  v8f acc0 = {}, acc1 = {};
  for (int kc = 0; kc < 512; kc += 32) {
    v16bf a0, a1, bb;
#pragma unroll
    for (int i = 0; i < 16; ++i) {
      int k = kc + kb + (i < 8 ? i : i + 8);  // ISA 16-bit A-fragment layout
      a0[i] = bfbits(a0p[k]);
      a1[i] = bfbits(a1p[k]);
      bb[i] = bfbits(bp[k]);
    }
    acc0 = wmma_bf16(a0, bb, acc0);
    acc1 = wmma_bf16(a1, bb, acc1);
  }
  int ncol = lane & 15;
  int mb   = (lane >> 4) << 3;
#pragma unroll
  for (int r = 0; r < 8; ++r) {
    int m = mb + r;
    D[(size_t)(m0 + m)      * 512 + n0 + ncol] = acc0[r];
    D[(size_t)(m0 + 16 + m) * 512 + n0 + ncol] = acc1[r];
  }
}

// ----------------------------------------------------------------- LSTM cell
__global__ void k_lstm(const float* __restrict__ emb, const int* __restrict__ dec,
                       const float* __restrict__ Wih, const float* __restrict__ Whh,
                       const float* __restrict__ bih, const float* __restrict__ bhh,
                       float* __restrict__ h, float* __restrict__ c,
                       float* __restrict__ buf, int t) {
  int b = blockIdx.x, tid = threadIdx.x;
  __shared__ float xs[EMBD], hs[HID], gs[G4];
  int tok = dec[b * TRGLEN + t];
  if (tok < 0 || tok >= NVOCAB) tok = UNKTOK;
  for (int k = tid; k < EMBD; k += 256) {
    xs[k] = emb[(size_t)tok * EMBD + k];
    hs[k] = h[b * HID + k];
  }
  __syncthreads();
  for (int jj = 0; jj < 8; ++jj) {
    int j = jj * 256 + tid;
    const float* wi = Wih + (size_t)j * EMBD;
    const float* wh = Whh + (size_t)j * HID;
    float g = bih[j] + bhh[j];
    for (int k = 0; k < EMBD; ++k) g += xs[k] * wi[k];
    for (int k = 0; k < HID;  ++k) g += hs[k] * wh[k];
    gs[j] = g;
  }
  __syncthreads();
  for (int u = tid; u < HID; u += 256) {
    float ig = gs[u], fg = gs[HID + u], gg = gs[2 * HID + u], og = gs[3 * HID + u];
    float cn = sigm(fg) * c[b * HID + u] + sigm(ig) * tanhf(gg);
    float hn = sigm(og) * tanhf(cn);
    c[b * HID + u] = cn;
    h[b * HID + u] = hn;
    buf[((size_t)b * TSTEPS + t) * HID + u] = hn;   // slot t unread this step (mask t'<t)
  }
}

// ------------------------------------------ attention + self-attention ctx
__global__ void k_attn(const float* __restrict__ h, const float* __restrict__ encproj,
                       const float* __restrict__ enc, const float* __restrict__ buf,
                       float* __restrict__ aw, float* __restrict__ ed,
                       float* __restrict__ ds, int t) {
  int b = blockIdx.x, tid = threadIdx.x;
  __shared__ float hs[HID];
  __shared__ float sc[SRCLEN];
  __shared__ float red[256];
  __shared__ float sw[TSTEPS];
  const float scale = 0.044194173824159216f;   // 1/sqrt(512)
  for (int k = tid; k < HID; k += 256) hs[k] = h[b * HID + k];
  __syncthreads();
  for (int s = tid; s < SRCLEN; s += 256) {
    const float* ep = encproj + ((size_t)b * SRCLEN + s) * HID;
    float d = 0.f;
    for (int k = 0; k < HID; ++k) d += hs[k] * ep[k];
    sc[s] = d * scale;
  }
  __syncthreads();
  float lm = -1e30f;
  for (int s = tid; s < SRCLEN; s += 256) lm = fmaxf(lm, sc[s]);
  red[tid] = lm; __syncthreads();
  for (int o = 128; o > 0; o >>= 1) { if (tid < o) red[tid] = fmaxf(red[tid], red[tid + o]); __syncthreads(); }
  float mx = red[0]; __syncthreads();
  float ls = 0.f;
  for (int s = tid; s < SRCLEN; s += 256) { float e = __expf(sc[s] - mx); sc[s] = e; ls += e; }
  red[tid] = ls; __syncthreads();
  for (int o = 128; o > 0; o >>= 1) { if (tid < o) red[tid] += red[tid + o]; __syncthreads(); }
  float inv = 1.0f / red[0]; __syncthreads();
  for (int s = tid; s < SRCLEN; s += 256) { sc[s] *= inv; aw[b * SRCLEN + s] = sc[s]; }
  __syncthreads();
  for (int u = tid; u < HID; u += 256) {
    float e = 0.f;
    for (int s = 0; s < SRCLEN; ++s) e += sc[s] * enc[((size_t)b * SRCLEN + s) * HID + u];
    ed[b * HID + u] = e;
  }
  if (t > 0) {
    if (tid < t) {
      const float* bp = buf + ((size_t)b * TSTEPS + tid) * HID;
      float d = 0.f;
      for (int k = 0; k < HID; ++k) d += hs[k] * bp[k];
      sw[tid] = d * scale;
    }
    __syncthreads();
    if (tid == 0) {
      float m2 = -1e30f;
      for (int i = 0; i < t; ++i) m2 = fmaxf(m2, sw[i]);
      float s2 = 0.f;
      for (int i = 0; i < t; ++i) { sw[i] = __expf(sw[i] - m2); s2 += sw[i]; }
      float in2 = 1.0f / s2;
      for (int i = 0; i < t; ++i) sw[i] *= in2;
    }
    __syncthreads();
    for (int u = tid; u < HID; u += 256) {
      float a = 0.f;
      for (int i = 0; i < t; ++i) a += sw[i] * buf[((size_t)b * TSTEPS + i) * HID + u];
      ds[b * HID + u] = a;
    }
  } else {
    for (int u = tid; u < HID; u += 256) ds[b * HID + u] = 0.f;
  }
}

// ------------------- feat = [h|ed|ds] -> bf16, copy-gate p, zero merged tail
__global__ void k_feat(const float* __restrict__ h, const float* __restrict__ ed,
                       const float* __restrict__ ds, const float* __restrict__ Wc,
                       const float* __restrict__ bc, unsigned short* __restrict__ featbf,
                       float* __restrict__ p, float* __restrict__ merged) {
  int b = blockIdx.x, tid = threadIdx.x;
  __shared__ float red[256];
  float acc = 0.f;
  for (int j = tid; j < KFEAT; j += 256) {
    float f = (j < HID) ? h[b * HID + j]
            : (j < 2 * HID ? ed[b * HID + j - HID] : ds[b * HID + j - 2 * HID]);
    featbf[(size_t)b * KFEAT + j] = f2bf(f);
    acc += f * Wc[j];
  }
  red[tid] = acc; __syncthreads();
  for (int o = 128; o > 0; o >>= 1) { if (tid < o) red[tid] += red[tid + o]; __syncthreads(); }
  if (tid == 0) p[b] = 1.0f / (1.0f + __expf(-(red[0] + bc[0])));
  for (int v = NVOCAB + tid; v < NVEXT; v += 256) merged[(size_t)b * NVEXT + v] = 0.f;
}

// ------------------- merged[:,0:32000] = (feat@Wv^T + b) * (1-p)  [WMMA]
__global__ void k_vocab_gemm(const unsigned short* __restrict__ Wv,    // [32000 x 1536] bf16
                             const unsigned short* __restrict__ F,     // [32 x 1536] bf16
                             const float* __restrict__ bvoc,
                             const float* __restrict__ p,
                             float* __restrict__ merged) {              // [32 x NVEXT]
  int wave = threadIdx.x >> 5, lane = threadIdx.x & 31;
  int nt = blockIdx.x * 8 + wave;                 // 0..1999
  int n0 = nt * 16;
  int row = lane & 15;
  int kb  = (lane >> 4) << 3;
  const unsigned short* a0p = F  + (size_t)row        * KFEAT;
  const unsigned short* a1p = F  + (size_t)(row + 16) * KFEAT;
  const unsigned short* bp  = Wv + (size_t)(n0 + row) * KFEAT;
  v8f acc0 = {}, acc1 = {};
  for (int kc = 0; kc < KFEAT; kc += 32) {
    v16bf a0, a1, bb;
#pragma unroll
    for (int i = 0; i < 16; ++i) {
      int k = kc + kb + (i < 8 ? i : i + 8);
      a0[i] = bfbits(a0p[k]);
      a1[i] = bfbits(a1p[k]);
      bb[i] = bfbits(bp[k]);
    }
    acc0 = wmma_bf16(a0, bb, acc0);   // rows 0..15 of batch
    acc1 = wmma_bf16(a1, bb, acc1);   // rows 16..31
  }
  int ncol = lane & 15;
  int mb   = (lane >> 4) << 3;
  float bias = bvoc[n0 + ncol];
#pragma unroll
  for (int r = 0; r < 8; ++r) {
    int m = mb + r;
    merged[(size_t)m        * NVEXT + n0 + ncol] = (acc0[r] + bias) * (1.0f - p[m]);
    merged[(size_t)(m + 16) * NVEXT + n0 + ncol] = (acc1[r] + bias) * (1.0f - p[m + 16]);
  }
}

// ------------------------------------------ copy distribution scatter-add
__global__ void k_scatter(const int* __restrict__ ext, const float* __restrict__ p,
                          const float* __restrict__ aw, float* __restrict__ merged) {
  int idx = blockIdx.x * 256 + threadIdx.x;
  if (idx >= BATCH * SRCLEN) return;
  int b = idx / SRCLEN;
  int v = ext[idx];
  if (v < 0) v = 0;
  if (v >= NVEXT) v = NVEXT - 1;
  atomicAdd(&merged[(size_t)b * NVEXT + v], p[b] * aw[idx]);
}

// ------------------------------------------ log-softmax + argmax -> d_out
__global__ void k_logsoftmax(const float* __restrict__ merged,
                             float* __restrict__ logp, float* __restrict__ preds, int t) {
  int b = blockIdx.x, tid = threadIdx.x;
  __shared__ float rv[256];
  __shared__ int   ri[256];
  const float* row = merged + (size_t)b * NVEXT;
  float lm = -1e30f; int li = 0;
  for (int v = tid; v < NVEXT; v += 256) {
    float x = row[v];
    if (x > lm) { lm = x; li = v; }
  }
  rv[tid] = lm; ri[tid] = li; __syncthreads();
  for (int o = 128; o > 0; o >>= 1) {
    if (tid < o) {
      if (rv[tid + o] > rv[tid] || (rv[tid + o] == rv[tid] && ri[tid + o] < ri[tid])) {
        rv[tid] = rv[tid + o]; ri[tid] = ri[tid + o];
      }
    }
    __syncthreads();
  }
  float mx = rv[0]; int arg = ri[0]; __syncthreads();
  float ls = 0.f;
  for (int v = tid; v < NVEXT; v += 256) ls += __expf(row[v] - mx);
  rv[tid] = ls; __syncthreads();
  for (int o = 128; o > 0; o >>= 1) { if (tid < o) rv[tid] += rv[tid + o]; __syncthreads(); }
  float lse = mx + __logf(rv[0]);
  float* out = logp + ((size_t)b * TSTEPS + t) * NVEXT;
  for (int v = tid; v < NVEXT; v += 256) out[v] = row[v] - lse;
  if (tid == 0) preds[b * TSTEPS + t] = (float)arg;
}

// ============================================================================
extern "C" void kernel_launch(void* const* d_in, const int* in_sizes, int n_in,
                              void* d_out, int out_size, void* d_ws, size_t ws_size,
                              hipStream_t stream) {
  (void)in_sizes; (void)n_in; (void)out_size; (void)ws_size;
  const int*   dec_inputs = (const int*)  d_in[0];
  const int*   enc_ext    = (const int*)  d_in[1];
  const float* enc_out    = (const float*)d_in[2];
  const float* enc_h0     = (const float*)d_in[3];
  const float* enc_c0     = (const float*)d_in[4];
  const float* emb        = (const float*)d_in[5];
  const float* W_ih       = (const float*)d_in[6];
  const float* W_hh       = (const float*)d_in[7];
  const float* b_ih       = (const float*)d_in[8];
  const float* b_hh       = (const float*)d_in[9];
  const float* W_attn     = (const float*)d_in[10];
  const float* W_vocab    = (const float*)d_in[11];
  const float* b_vocab    = (const float*)d_in[12];
  const float* W_copy     = (const float*)d_in[13];
  const float* b_copy     = (const float*)d_in[14];

  float* logps = (float*)d_out;
  float* preds = logps + (size_t)BATCH * TSTEPS * NVEXT;

  // workspace carve-out (256B aligned)
  char* ws = (char*)d_ws;
  size_t off = 0;
  auto carve = [&](size_t bytes) { size_t o = off; off = (off + bytes + 255) & ~(size_t)255; return o; };
  unsigned short* wv_bf   = (unsigned short*)(ws + carve((size_t)NVOCAB * KFEAT * 2));
  unsigned short* enc_bf  = (unsigned short*)(ws + carve((size_t)BATCH * SRCLEN * HID * 2));
  unsigned short* wat_bf  = (unsigned short*)(ws + carve((size_t)HID * HID * 2));
  float* enc_proj = (float*)(ws + carve((size_t)BATCH * SRCLEN * HID * 4));
  float* h        = (float*)(ws + carve((size_t)BATCH * HID * 4));
  float* c        = (float*)(ws + carve((size_t)BATCH * HID * 4));
  float* buf      = (float*)(ws + carve((size_t)BATCH * TSTEPS * HID * 4));
  float* aw       = (float*)(ws + carve((size_t)BATCH * SRCLEN * 4));
  float* ed       = (float*)(ws + carve((size_t)BATCH * HID * 4));
  float* ds       = (float*)(ws + carve((size_t)BATCH * HID * 4));
  unsigned short* feat_bf = (unsigned short*)(ws + carve((size_t)BATCH * KFEAT * 2));
  float* pgate    = (float*)(ws + carve(256));
  float* merged   = (float*)(ws + carve((size_t)BATCH * NVEXT * 4));

  // one-time: bf16 conversions + enc_proj GEMM + state init
  {
    int n = NVOCAB * KFEAT;
    k_f32_to_bf16<<<(n + 255) / 256, 256, 0, stream>>>(W_vocab, wv_bf, n);
  }
  {
    int n = BATCH * SRCLEN * HID;
    k_f32_to_bf16<<<(n + 255) / 256, 256, 0, stream>>>(enc_out, enc_bf, n);
  }
  {
    int n = HID * HID;
    k_f32_to_bf16<<<(n + 255) / 256, 256, 0, stream>>>(W_attn, wat_bf, n);
  }
  k_encproj_gemm<<<1600, 256, 0, stream>>>(enc_bf, wat_bf, enc_proj);
  hipMemcpyAsync(h, enc_h0, (size_t)BATCH * HID * 4, hipMemcpyDeviceToDevice, stream);
  hipMemcpyAsync(c, enc_c0, (size_t)BATCH * HID * 4, hipMemcpyDeviceToDevice, stream);

  // sequential decode
  for (int t = 0; t < TSTEPS; ++t) {
    k_lstm<<<BATCH, 256, 0, stream>>>(emb, dec_inputs, W_ih, W_hh, b_ih, b_hh, h, c, buf, t);
    k_attn<<<BATCH, 256, 0, stream>>>(h, enc_proj, enc_out, buf, aw, ed, ds, t);
    k_feat<<<BATCH, 256, 0, stream>>>(h, ed, ds, W_copy, b_copy, feat_bf, pgate, merged);
    k_vocab_gemm<<<250, 256, 0, stream>>>(wv_bf, feat_bf, b_vocab, pgate, merged);
    k_scatter<<<(BATCH * SRCLEN + 255) / 256, 256, 0, stream>>>(enc_ext, pgate, aw, merged);
    k_logsoftmax<<<BATCH, 256, 0, stream>>>(merged, logps, preds, t);
  }
}
